// ImprovedSentenceGNN_13889924235781
// MI455X (gfx1250) — compile-verified, matrix operand order checked by reading, not run on hardware
//
#include <hip/hip_runtime.h>
#include <hip/hip_bf16.h>

// ============================================================================
// ImprovedSentenceGNN on MI455X (gfx1250)
//  - dense projections via v_wmma_f32_16x16x32_bf16, 128x128x32 block tiles,
//    8 waves/block, 4x2 16x16 f32 accum tiles per wave, double-buffered LDS,
//    global->reg->LDS software pipeline + global_prefetch of s+2 tiles.
//    Weights are packed PRE-TRANSPOSED so both A and Bt stage with b128 ops.
//  - edge phase: wave32-per-edge gather + LeakyReLU + attention dot,
//    segment softmax with fp32/ordered-int atomics, scatter-add aggregation
//  - LN+GELU / pooling / head as block-per-row kernels
// ============================================================================

#define WAVE 32

typedef __attribute__((ext_vector_type(16))) __bf16 v16bf;
typedef __attribute__((ext_vector_type(8)))  float  v8f;

__device__ __forceinline__ unsigned short f2bf(float f) {
  unsigned int u = __float_as_uint(f);
  u = (u + 0x7FFFu + ((u >> 16) & 1u)) >> 16;   // round-to-nearest-even
  return (unsigned short)u;
}

// order-preserving float->int key for atomicMax on float values
__device__ __forceinline__ int fkey(float f) {
  int i = __float_as_int(f);
  return i >= 0 ? i : (i ^ 0x7fffffff);
}
__device__ __forceinline__ float unfkey(int i) {
  return __int_as_float(i >= 0 ? i : (i ^ 0x7fffffff));
}

// ---------------------------------------------------------------------------
// fills / casts / packs
// ---------------------------------------------------------------------------
__global__ void k_fill_f32(float* p, float v, size_t n) {
  size_t i = (size_t)blockIdx.x * 256 + threadIdx.x;
  if (i < n) p[i] = v;
}
__global__ void k_fill_i32(int* p, int v, size_t n) {
  size_t i = (size_t)blockIdx.x * 256 + threadIdx.x;
  if (i < n) p[i] = v;
}

// row-wise f32 -> bf16 cast with independent strides (handles J-slice input)
__global__ __launch_bounds__(256)
void k_cast_bf16(const float* __restrict__ src, size_t sstride,
                 unsigned short* __restrict__ dst, size_t dstride, int cols) {
  size_t r = blockIdx.x;
  const float* s = src + r * sstride;
  unsigned short* d = dst + r * dstride;
  for (int c = threadIdx.x; c < cols; c += 256) d[c] = f2bf(s[c]);
}

// cast + transpose a [K x Nc] f32 weight into bf16 [Nc x K]
__global__ __launch_bounds__(256)
void k_cast_w_t(const float* __restrict__ src, int K, int Nc,
                unsigned short* __restrict__ dst) {
  int k = blockIdx.x;                       // 0..K-1
  for (int c = threadIdx.x; c < Nc; c += 256)
    dst[(size_t)c * K + k] = f2bf(src[(size_t)k * Nc + c]);
}

// pack [Wl | Wr] (each [768x768]) -> bf16 TRANSPOSED [1536 x 768]
__global__ __launch_bounds__(256)
void k_pack_wlr_t(const float* __restrict__ Wl, const float* __restrict__ Wr,
                  unsigned short* __restrict__ Wt) {
  int k = blockIdx.x;   // 0..767
  for (int c = threadIdx.x; c < 1536; c += 256) {
    float v = (c < 768) ? Wl[(size_t)k * 768 + c] : Wr[(size_t)k * 768 + (c - 768)];
    Wt[(size_t)c * 768 + k] = f2bf(v);
  }
}
__global__ void k_pack_bias(const float* __restrict__ bl, const float* __restrict__ br,
                            float* __restrict__ bias) {
  for (int c = threadIdx.x; c < 1536; c += 256)
    bias[c] = (c < 768) ? bl[c] : br[c - 768];
}

// ---------------------------------------------------------------------------
// bf16 WMMA GEMM: Out[M x Nc] = A[M x K] * B[K x Nc] + bias, B given as
// Bt[Nc x K] (pre-transposed). Block tile 128x128, K-step 32, 8 waves
// (2 along M x 4 along N), 4x2 accum tiles per wave. Double-buffered LDS
// with a global->register->LDS pipeline; s+2 tiles prefetched into L2.
// ---------------------------------------------------------------------------
#define TM 128
#define TN 128
#define TK 32
#define LSTR 40   // ushort LDS row stride (80B, 16B-aligned, bank-conflict pad)

__global__ __launch_bounds__(256)
void k_gemm_bf16(const unsigned short* __restrict__ A, int lda,
                 const unsigned short* __restrict__ Bt, int ldbt,
                 const float* __restrict__ bias,
                 float* __restrict__ Out, int ldo,
                 int M, int Nc, int K) {
  __shared__ unsigned short As[2][TM * LSTR];
  __shared__ unsigned short Bs[2][TN * LSTR];

  const int tid  = threadIdx.x;
  const int wid  = tid >> 5;
  const int lane = tid & 31;
  const int bm = blockIdx.x * TM;
  const int bn = blockIdx.y * TN;
  const int waveM = (wid & 1) * 64;   // 2 waves along M
  const int waveN = (wid >> 1) * 32;  // 4 waves along N

  // staging: thread -> (row r, 16-elem half hf); same pattern for A and Bt
  const int r  = tid >> 1;
  const int hf = (tid & 1) * 16;
  const int aRow = bm + r;
  const bool aOk = (aRow < M);
  const unsigned short* gA = A  + (size_t)aRow     * lda  + hf;
  const unsigned short* gB = Bt + (size_t)(bn + r) * ldbt + hf;

  v8f acc[4][2] = {};
  const int rlo  = lane & 15;
  const int koff = (lane >> 4) * 8;   // fragment K-offset per ISA 16-bit layout
  const int nsteps = K / TK;

  // ---- prologue: stage step 0
  uint4 ra0 = {0,0,0,0}, ra1 = {0,0,0,0}, rb0, rb1;
  if (aOk) { const uint4* p = (const uint4*)gA; ra0 = p[0]; ra1 = p[1]; }
  { const uint4* p = (const uint4*)gB; rb0 = p[0]; rb1 = p[1]; }
  { uint4* q = (uint4*)(&As[0][r * LSTR + hf]); q[0] = ra0; q[1] = ra1; }
  { uint4* q = (uint4*)(&Bs[0][r * LSTR + hf]); q[0] = rb0; q[1] = rb1; }
  __syncthreads();

  for (int s = 0; s < nsteps; ++s) {
    const int buf = s & 1;

    // prefetch s+2 into L2 (global_prefetch_b8 path)
    if (s + 2 < nsteps) {
      __builtin_prefetch(gA + (size_t)(s + 2) * TK, 0, 1);
      __builtin_prefetch(gB + (size_t)(s + 2) * TK, 0, 1);
    }
    // issue global loads for s+1 (land in registers while WMMAs run)
    if (s + 1 < nsteps) {
      const int k0 = (s + 1) * TK;
      ra0 = ra1 = (uint4){0,0,0,0};
      if (aOk) { const uint4* p = (const uint4*)(gA + k0); ra0 = p[0]; ra1 = p[1]; }
      { const uint4* p = (const uint4*)(gB + k0); rb0 = p[0]; rb1 = p[1]; }
    }

    // ---- fragments from LDS[buf]
    v16bf a[4], b[2];
#pragma unroll
    for (int i = 0; i < 4; ++i) {
      const unsigned short* base = &As[buf][(waveM + i * 16 + rlo) * LSTR];
      uint4* av = (uint4*)&a[i];
      av[0] = *(const uint4*)(base + koff);
      av[1] = *(const uint4*)(base + koff + 16);
    }
#pragma unroll
    for (int j = 0; j < 2; ++j) {
      const unsigned short* base = &Bs[buf][(waveN + j * 16 + rlo) * LSTR];
      uint4* bv = (uint4*)&b[j];
      bv[0] = *(const uint4*)(base + koff);
      bv[1] = *(const uint4*)(base + koff + 16);
    }
#pragma unroll
    for (int i = 0; i < 4; ++i)
#pragma unroll
      for (int j = 0; j < 2; ++j)
        acc[i][j] = __builtin_amdgcn_wmma_f32_16x16x32_bf16(
            false, a[i], false, b[j], (short)0, acc[i][j], false, false);

    // ---- stash s+1 tiles into the other LDS buffer
    if (s + 1 < nsteps) {
      const int nb = buf ^ 1;
      uint4* qa = (uint4*)(&As[nb][r * LSTR + hf]); qa[0] = ra0; qa[1] = ra1;
      uint4* qb = (uint4*)(&Bs[nb][r * LSTR + hf]); qb[0] = rb0; qb[1] = rb1;
    }
    __syncthreads();
  }

  // ---- epilogue: C/D layout (VGPR v: lanes0-15 M=v, lanes16-31 M=8+v; N=lane&15)
  const int mhalf = (lane < 16) ? 0 : 8;
#pragma unroll
  for (int i = 0; i < 4; ++i) {
#pragma unroll
    for (int j = 0; j < 2; ++j) {
      int col = bn + waveN + j * 16 + rlo;
      float bv = bias ? bias[col] : 0.0f;
#pragma unroll
      for (int v = 0; v < 8; ++v) {
        int row = bm + waveM + i * 16 + mhalf + v;
        if (row < M) Out[(size_t)row * ldo + col] = acc[i][j][v] + bv;
      }
    }
  }
}

// ---------------------------------------------------------------------------
// edge phase: xlr is [N x 1536] f32 (cols 0..767 = xl, 768..1535 = xr)
// ---------------------------------------------------------------------------
__global__ __launch_bounds__(256)
void k_edge_logits(const float* __restrict__ xlr,
                   const long long* __restrict__ ei, int E_,
                   const float* __restrict__ att,
                   float* __restrict__ logits, int* __restrict__ mInt, int EP) {
  int wid = threadIdx.x >> 5, lane = threadIdx.x & 31;
  int e = blockIdx.x * 8 + wid;
  if (e >= EP) return;
  long long s, t;
  if (e < E_) { s = ei[e]; t = ei[E_ + e]; } else { s = t = (long long)(e - E_); }
  const float* xs = xlr + (size_t)s * 1536;
  const float* xt = xlr + (size_t)t * 1536 + 768;
#pragma unroll
  for (int h = 0; h < 8; ++h) {
    float sum = 0.f;
#pragma unroll
    for (int it = 0; it < 3; ++it) {
      int c = h * 96 + it * 32 + lane;
      float v = xs[c] + xt[c];
      v = v > 0.f ? v : 0.2f * v;      // LeakyReLU(0.2)
      sum += v * att[c];
    }
    for (int o = 16; o > 0; o >>= 1) sum += __shfl_xor(sum, o, WAVE);
    if (lane == 0) {
      logits[(size_t)e * 8 + h] = sum;
      atomicMax(&mInt[t * 8 + h], fkey(sum));
    }
  }
}

__global__ __launch_bounds__(256)
void k_edge_exp(const long long* __restrict__ ei, int E_, int EP,
                float* __restrict__ logits, const int* __restrict__ mInt,
                float* __restrict__ sSum) {
  int idx = blockIdx.x * 256 + threadIdx.x;
  if (idx >= EP * 8) return;
  int e = idx >> 3, h = idx & 7;
  long long t = (e < E_) ? ei[E_ + e] : (long long)(e - E_);
  float m  = unfkey(mInt[t * 8 + h]);
  float ex = expf(logits[idx] - m);
  logits[idx] = ex;                     // overwrite with exp
  atomicAdd(&sSum[t * 8 + h], ex);
}

__global__ __launch_bounds__(256)
void k_edge_agg(const float* __restrict__ xlr,
                const long long* __restrict__ ei, int E_, int EP,
                const float* __restrict__ ex, const float* __restrict__ sSum,
                float* __restrict__ agg) {
  int wid = threadIdx.x >> 5, lane = threadIdx.x & 31;
  int e = blockIdx.x * 8 + wid;
  if (e >= EP) return;
  long long s, t;
  if (e < E_) { s = ei[e]; t = ei[E_ + e]; } else { s = t = (long long)(e - E_); }
  float al = 0.f;
  if (lane < 8) {
    float sv = sSum[t * 8 + lane];
    al = ex[(size_t)e * 8 + lane] / (sv > 0.f ? sv : 1.f);
  }
  const float* xs = xlr + (size_t)s * 1536;
  float* o = agg + (size_t)t * 768;
#pragma unroll
  for (int it = 0; it < 24; ++it) {
    int c = it * 32 + lane;
    float a = __shfl(al, c / 96, WAVE);
    atomicAdd(&o[c], a * xs[c]);
  }
}

// ---------------------------------------------------------------------------
// residual + LayerNorm + GELU(exact), writes into JK slice
// ---------------------------------------------------------------------------
__device__ __forceinline__ float blockSum256(float v, volatile float* red) {
  for (int o = 16; o > 0; o >>= 1) v += __shfl_xor(v, o, WAVE);
  if ((threadIdx.x & 31) == 0) red[threadIdx.x >> 5] = v;
  __syncthreads();
  float t = 0.f;
#pragma unroll
  for (int i = 0; i < 8; ++i) t += red[i];
  __syncthreads();
  return t;
}

__global__ __launch_bounds__(256)
void k_ln_gelu(const float* __restrict__ agg,
               const float* __restrict__ hprev, size_t hstride,
               const float* __restrict__ gbias,
               const float* __restrict__ lw, const float* __restrict__ lb,
               float* __restrict__ outJ, size_t ostride) {
  __shared__ float red[8];
  size_t n = blockIdx.x;
  const float* ap = agg + n * 768;
  const float* hp = hprev + n * hstride;
  float vals[3], sum = 0.f;
#pragma unroll
  for (int i = 0; i < 3; ++i) {
    int c = i * 256 + threadIdx.x;
    float v = ap[c] + gbias[c] + hp[c];
    vals[i] = v;
    sum += v;
  }
  float mean = blockSum256(sum, red) * (1.f / 768.f);
  float vs = 0.f;
#pragma unroll
  for (int i = 0; i < 3; ++i) { float d = vals[i] - mean; vs += d * d; }
  float var = blockSum256(vs, red) * (1.f / 768.f);
  float inv = rsqrtf(var + 1e-5f);
#pragma unroll
  for (int i = 0; i < 3; ++i) {
    int c = i * 256 + threadIdx.x;
    float y = (vals[i] - mean) * inv * lw[c] + lb[c];
    float g = 0.5f * y * (1.f + erff(y * 0.70710678118654752f));
    outJ[n * ostride + c] = g;
  }
}

// ---------------------------------------------------------------------------
// pooling + head
// ---------------------------------------------------------------------------
__global__ void k_count(const long long* __restrict__ batch, float* __restrict__ cnt, int Nn) {
  int n = blockIdx.x * 256 + threadIdx.x;
  if (n < Nn) atomicAdd(&cnt[batch[n]], 1.0f);
}

__global__ __launch_bounds__(256)
void k_pool(const float* __restrict__ J, const long long* __restrict__ batch,
            float* __restrict__ P) {
  size_t n = blockIdx.x;
  int c = blockIdx.y * 256 + threadIdx.x;     // 9*256 == 2304
  long long g = batch[n];
  atomicAdd(&P[(size_t)g * 2304 + c], J[n * 2304 + c]);
}

__global__ __launch_bounds__(256)
void k_final(const float* __restrict__ P, const float* __restrict__ cnt,
             const float* __restrict__ lw, const float* __restrict__ lb,
             const float* __restrict__ Wout, const float* __restrict__ bout,
             float* __restrict__ out) {
  __shared__ float y[2304];
  __shared__ float red[8];
  __shared__ float pacc[8][32];
  int g = blockIdx.x;
  float c = cnt[g];
  float invc = 1.f / (c > 1.f ? c : 1.f);
  float vals[9], sum = 0.f;
#pragma unroll
  for (int i = 0; i < 9; ++i) {
    int k = i * 256 + threadIdx.x;
    float v = P[(size_t)g * 2304 + k] * invc;
    vals[i] = v;
    sum += v;
  }
  float mean = blockSum256(sum, red) * (1.f / 2304.f);
  float vs = 0.f;
#pragma unroll
  for (int i = 0; i < 9; ++i) { float d = vals[i] - mean; vs += d * d; }
  float var = blockSum256(vs, red) * (1.f / 2304.f);
  float inv = rsqrtf(var + 1e-5f);
#pragma unroll
  for (int i = 0; i < 9; ++i) {
    int k = i * 256 + threadIdx.x;
    y[k] = (vals[i] - mean) * inv * lw[k] + lb[k];
  }
  __syncthreads();
  // 2304x32 dot: 8 K-chunks x 32 classes across all 256 threads
  {
    int j  = threadIdx.x & 31;
    int ch = threadIdx.x >> 5;
    float acc = 0.f;
    for (int k = ch * 288; k < ch * 288 + 288; ++k)
      acc += y[k] * Wout[(size_t)k * 32 + j];
    pacc[ch][j] = acc;
  }
  __syncthreads();
  if (threadIdx.x < 32) {
    float t = bout[threadIdx.x];
#pragma unroll
    for (int i = 0; i < 8; ++i) t += pacc[i][threadIdx.x];
    out[(size_t)g * 32 + threadIdx.x] = t;
  }
}

// ===========================================================================
// host orchestration
// ===========================================================================
extern "C" void kernel_launch(void* const* d_in, const int* in_sizes, int n_in,
                              void* d_out, int out_size, void* d_ws, size_t ws_size,
                              hipStream_t stream) {
  const float*      x        = (const float*)d_in[0];
  const long long*  ei       = (const long long*)d_in[1];   // int64 in reference
  const long long*  batch    = (const long long*)d_in[2];
  const float*      w_in     = (const float*)d_in[3];
  const float*      b_in     = (const float*)d_in[4];
  const float*      Wl       = (const float*)d_in[5];
  const float*      bl       = (const float*)d_in[6];
  const float*      Wr       = (const float*)d_in[7];
  const float*      br       = (const float*)d_in[8];
  const float*      att      = (const float*)d_in[9];
  const float*      gat_bias = (const float*)d_in[10];
  const float*      ln_w     = (const float*)d_in[11];
  const float*      ln_b     = (const float*)d_in[12];
  const float*      out_ln_w = (const float*)d_in[13];
  const float*      out_ln_b = (const float*)d_in[14];
  const float*      w_out    = (const float*)d_in[15];
  const float*      b_out    = (const float*)d_in[16];

  const int Nn = in_sizes[2];            // 20000
  const int E_ = in_sizes[1] / 2;        // 160000
  const int EP = E_ + Nn;                // edges incl. self loops
  const int G  = out_size / 32;          // 256

  // ---- workspace layout (256B aligned slabs)
  char* base = (char*)d_ws;
  size_t off = 0;
  auto take = [&](size_t bytes) -> char* {
    char* p = base + off;
    off += (bytes + 255) & ~(size_t)255;
    return p;
  };
  float*          h0    = (float*)take((size_t)Nn * 768 * 4);
  unsigned short* hbf   = (unsigned short*)take((size_t)Nn * 768 * 2);
  unsigned short* wbf   = (unsigned short*)take((size_t)768 * 1536 * 2);  // transposed weights
  float*          bias2 = (float*)take(1536 * 4);
  float*          xlr   = (float*)take((size_t)Nn * 1536 * 4);
  float*          lgts  = (float*)take((size_t)EP * 8 * 4);
  int*            mInt  = (int*)take((size_t)Nn * 8 * 4);
  float*          sSum  = (float*)take((size_t)Nn * 8 * 4);
  float*          agg   = (float*)take((size_t)Nn * 768 * 4);
  float*          J     = (float*)take((size_t)Nn * 2304 * 4);
  float*          P     = (float*)take((size_t)G * 2304 * 4);
  float*          cnt   = (float*)take((size_t)G * 4);

  const int mtiles = (Nn + TM - 1) / TM;

  // ---- h0 = x @ w_in + b_in
  k_cast_bf16<<<Nn, 256, 0, stream>>>(x, 768, hbf, 768, 768);
  k_cast_w_t<<<768, 256, 0, stream>>>(w_in, 768, 768, wbf);
  {
    dim3 grid(mtiles, 768 / TN);
    k_gemm_bf16<<<grid, 256, 0, stream>>>(hbf, 768, wbf, 768, b_in, h0, 768,
                                          Nn, 768, 768);
  }

  for (int l = 0; l < 3; ++l) {
    const float* hsrc = (l == 0) ? h0 : (J + (size_t)(l - 1) * 768);
    size_t hstride = (l == 0) ? 768 : 2304;

    // projections: xlr = h @ [Wl|Wr] + [bl|br]
    k_cast_bf16<<<Nn, 256, 0, stream>>>(hsrc, hstride, hbf, 768, 768);
    k_pack_wlr_t<<<768, 256, 0, stream>>>(Wl + (size_t)l * 768 * 768,
                                          Wr + (size_t)l * 768 * 768, wbf);
    k_pack_bias<<<1, 256, 0, stream>>>(bl + (size_t)l * 768, br + (size_t)l * 768, bias2);
    {
      dim3 grid(mtiles, 1536 / TN);
      k_gemm_bf16<<<grid, 256, 0, stream>>>(hbf, 768, wbf, 768, bias2, xlr, 1536,
                                            Nn, 1536, 768);
    }

    // segment softmax state
    k_fill_i32<<<(Nn * 8 + 255) / 256, 256, 0, stream>>>(mInt, (int)0x80000000, (size_t)Nn * 8);
    k_fill_f32<<<(Nn * 8 + 255) / 256, 256, 0, stream>>>(sSum, 0.f, (size_t)Nn * 8);
    k_fill_f32<<<((size_t)Nn * 768 + 255) / 256, 256, 0, stream>>>(agg, 0.f, (size_t)Nn * 768);

    k_edge_logits<<<(EP + 7) / 8, 256, 0, stream>>>(xlr, ei, E_, att + (size_t)l * 768,
                                                    lgts, mInt, EP);
    k_edge_exp<<<(EP * 8 + 255) / 256, 256, 0, stream>>>(ei, E_, EP, lgts, mInt, sSum);
    k_edge_agg<<<(EP + 7) / 8, 256, 0, stream>>>(xlr, ei, E_, EP, lgts, sSum, agg);

    // h_{l+1} = gelu(LN(agg + gat_bias + h_l)) -> J slice l
    k_ln_gelu<<<Nn, 256, 0, stream>>>(agg, hsrc, hstride, gat_bias + (size_t)l * 768,
                                      ln_w + (size_t)l * 768, ln_b + (size_t)l * 768,
                                      J + (size_t)l * 768, 2304);
  }

  // ---- global mean pool + out LN + head
  k_fill_f32<<<((size_t)G * 2304 + 255) / 256, 256, 0, stream>>>(P, 0.f, (size_t)G * 2304);
  k_fill_f32<<<(G + 255) / 256, 256, 0, stream>>>(cnt, 0.f, (size_t)G);
  k_count<<<(Nn + 255) / 256, 256, 0, stream>>>(batch, cnt, Nn);
  {
    dim3 grid(Nn, 9);
    k_pool<<<grid, 256, 0, stream>>>(J, batch, P);
  }
  k_final<<<G, 256, 0, stream>>>(P, cnt, out_ln_w, out_ln_b, w_out, b_out, (float*)d_out);

  (void)n_in; (void)ws_size;
}
